// DotProductAttention_89481348645697
// MI455X (gfx1250) — matrix operand order repaired and checked by reading, not compiled, not run on hardware
//
#include <hip/hip_runtime.h>

#define B_ 8
#define L_ 2048
#define D_ 1024

typedef __attribute__((ext_vector_type(16))) __bf16 v16bf;
typedef __attribute__((ext_vector_type(8)))  __bf16 v8bf;
typedef __attribute__((ext_vector_type(4)))  __bf16 v4bf;
typedef __attribute__((ext_vector_type(8)))  float  v8f;
typedef __attribute__((ext_vector_type(4)))  unsigned int v4u;
typedef __attribute__((ext_vector_type(8)))  int v8i;
typedef __attribute__((ext_vector_type(4)))  int v4i;

struct B32x8 { uint4 lo, hi; };
struct P2    { v8bf a, b; };

// LDS double-buffer layout (bytes). K tile: 32x1024 bf16, 16B pad per 1024B -> 2080B/row.
// V tile: 1024x32 bf16, 16B pad per 64B row -> 80B/row (conflict-free b128 reads).
#define KBYTES 66560
#define VBYTES 81920
#define KOFF(i) ((unsigned)(i) * KBYTES)
#define VOFF(i) (2u * KBYTES + (unsigned)(i) * VBYTES)
#define DYN_LDS (2u * KBYTES + 2u * VBYTES)   // 296960 B

static __device__ inline v16bf load_bf16x16(const void* p) {
    B32x8 raw;
    raw.lo = *(const uint4*)(p);
    raw.hi = *(const uint4*)((const char*)p + 16);
    return __builtin_bit_cast(v16bf, raw);
}

static __device__ inline v8f wmma_bf16(v16bf a, v16bf b, v8f c) {
    return __builtin_amdgcn_wmma_f32_16x16x32_bf16(false, a, false, b, (short)0, c, false, false);
}

static __device__ inline float rmax16(float v) {
    #pragma unroll
    for (int off = 8; off; off >>= 1) v = fmaxf(v, __shfl_xor(v, off, 16));
    return v;
}
static __device__ inline float rsum16(float v) {
    #pragma unroll
    for (int off = 8; off; off >>= 1) v += __shfl_xor(v, off, 16);
    return v;
}

// Issue one 2D TDM tile load (global -> LDS). Dims/strides in elements (data_size = 2B).
static __device__ inline void tdm_load_2d(unsigned lds_off, const void* gptr,
                                          unsigned tensor_d0, unsigned tensor_d1,
                                          unsigned tile_d0, unsigned tile_d1,
                                          unsigned stride0,
                                          unsigned pad_interval, unsigned pad_amount) {
    unsigned long long ga = (unsigned long long)(uintptr_t)gptr;
    v4u g0;
    g0[0] = 1u;                                            // count=1, user descriptor
    g0[1] = lds_off;                                       // LDS byte address
    g0[2] = (unsigned)ga;                                  // global_addr[31:0]
    g0[3] = (unsigned)((ga >> 32) & 0x1FFFFFFu) | (2u << 30); // global_addr[56:32] | type=2
    v8i g1;
    g1[0] = (int)((1u << 16)                               // data_size = 2 bytes
                | (1u << 20)                               // pad_enable
                | (pad_interval << 22)
                | (pad_amount << 25));
    g1[1] = (int)((tensor_d0 & 0xFFFFu) << 16);            // tensor_dim0[15:0] @ bits 63:48
    g1[2] = (int)((tensor_d0 >> 16) | ((tensor_d1 & 0xFFFFu) << 16));
    g1[3] = (int)((tensor_d1 >> 16) | (tile_d0 << 16));    // tile_dim0 @ bits 127:112
    g1[4] = (int)(tile_d1 & 0xFFFFu);                      // tile_dim1, tile_dim2=0
    g1[5] = (int)stride0;                                  // tensor_dim0_stride[31:0]
    g1[6] = 0;
    g1[7] = 0;
    v4i zz4 = (v4i)(0);
    v8i zz8 = (v8i)(0);
    __builtin_amdgcn_tensor_load_to_lds(g0, g1, zz4, zz4, zz8, 0);
}

// ---------------- pre-pass: K fp32 -> bf16 (row major) ----------------
__global__ void cvt_k_kernel(const float* __restrict__ k, __bf16* __restrict__ kbf, int n4) {
    int i = blockIdx.x * blockDim.x + threadIdx.x;
    if (i < n4) {
        float4 f = ((const float4*)k)[i];
        v4bf o;
        o[0] = (__bf16)f.x; o[1] = (__bf16)f.y; o[2] = (__bf16)f.z; o[3] = (__bf16)f.w;
        ((v4bf*)kbf)[i] = o;
    }
}

// ---------------- pre-pass: V fp32 [B][L][D] -> bf16 transposed [B][D][L] ----------------
__global__ void transpose_v_kernel(const float* __restrict__ v, __bf16* __restrict__ vt) {
    __shared__ float tile[32][33];
    int b  = blockIdx.z;
    int l0 = blockIdx.x * 32;
    int d0 = blockIdx.y * 32;
    int tx = threadIdx.x, ty = threadIdx.y;
    const float* src = v + ((size_t)(b * L_ + l0)) * D_ + d0;
    #pragma unroll
    for (int j = 0; j < 32; j += 8)
        tile[ty + j][tx] = src[(size_t)(ty + j) * D_ + tx];
    __syncthreads();
    __bf16* dst = vt + ((size_t)(b * D_ + d0)) * L_ + l0;
    #pragma unroll
    for (int j = 0; j < 32; j += 8)
        dst[(size_t)(ty + j) * L_ + tx] = (__bf16)tile[tx][ty + j];
}

// -------- fused flash attention: M=32 q rows/WG, 8 waves (2 q-subtiles x 4 D-slices),
// -------- TDM double-buffered K/V tiles in LDS --------
__global__ __launch_bounds__(256) void attn_kernel(const float* __restrict__ q,
                                                   const __bf16* __restrict__ kbf,
                                                   const __bf16* __restrict__ vt,
                                                   const int*   __restrict__ rep_mask,
                                                   float*       __restrict__ out) {
    extern __shared__ char smem[];                       // dynamic: K0,K1,V0,V1 (TDM dest)
    __shared__ float sred[2][2][4][16][16];              // [qs][nt][ws][row][col]
    __shared__ alignas(16) __bf16 pbuf[8][16][32];       // per-wave P tile

    const int lane = threadIdx.x & 31;
    const int wave = threadIdx.x >> 5;   // 0..7
    const int qs   = wave >> 2;          // q-subtile
    const int ws   = wave & 3;           // D-slice
    const int hi   = lane >> 4;
    const int col  = lane & 15;
    const int b    = blockIdx.y;
    const int q0   = blockIdx.x * 32 + qs * 16;
    const int woff = ws * 256;

    const unsigned ldsbase = (unsigned)(uintptr_t)smem;

    // ---- preload Q A-fragments (8 frags of 16x32 over this wave's D slice) ----
    v16bf qa[8];
    {
        const float* qrow = q + ((size_t)(b * L_ + q0 + col)) * D_ + woff;
        const int c0 = hi * 16, c1 = hi * 16 + 8;
        #pragma unroll
        for (int f = 0; f < 8; ++f) {
            const float* qp = qrow + f * 32;
            v16bf a;
            #pragma unroll
            for (int i = 0; i < 8; ++i) a[i]     = (__bf16)qp[c0 + i];
            #pragma unroll
            for (int i = 0; i < 8; ++i) a[8 + i] = (__bf16)qp[c1 + i];
            qa[f] = a;
        }
    }

    v8f o_acc[16];
    #pragma unroll
    for (int t = 0; t < 16; ++t) o_acc[t] = (v8f)(0.0f);
    float m_run[8], l_run[8];
    #pragma unroll
    for (int r = 0; r < 8; ++r) { m_run[r] = 0.0f; l_run[r] = 0.0f; }

    const __bf16* kbase = kbf + (size_t)b * L_ * D_;
    const __bf16* vbase = vt  + (size_t)b * D_ * L_;

    // prime buffer 0
    if (wave == 0) {
        tdm_load_2d(ldsbase + KOFF(0), kbase,            D_, L_, D_, 32,  D_, 7u, 3u);
        tdm_load_2d(ldsbase + VOFF(0), vbase,            L_, D_, 32, D_,  L_, 3u, 3u);
    }

    // byte offset of this wave's fragment chunk within a padded K row
    const int kdb  = (woff + hi * 16) * 2;

    for (int it = 0; it < 64; ++it) {
        const int kb  = it * 32;
        const int cur = it & 1;

        if (wave == 0) __builtin_amdgcn_s_wait_tensorcnt(0);
        __syncthreads();                                 // buf[cur] ready; buf[cur^1] free
        if (wave == 0 && it + 1 < 64) {
            const int kn = kb + 32;
            tdm_load_2d(ldsbase + KOFF(cur ^ 1), kbase + (size_t)kn * D_, D_, L_, D_, 32, D_, 7u, 3u);
            tdm_load_2d(ldsbase + VOFF(cur ^ 1), vbase + kn,              L_, D_, 32, D_, L_, 3u, 3u);
        }

        // issue the key-mask loads early so global latency hides under the WMMA stream
        const int mk0i = rep_mask[(size_t)b * L_ + kb + col];
        const int mk1i = rep_mask[(size_t)b * L_ + kb + 16 + col];

        // ---- partial S = Q(slice) . K(slice)^T from LDS K tile (pipelined frags) ----
        const char* ktile = smem + KOFF(cur);
        const char* krow0 = ktile + (col)      * 2080;
        const char* krow1 = ktile + (16 + col) * 2080;
        v8f s0 = (v8f)(0.0f), s1 = (v8f)(0.0f);
        {
            int db  = kdb;
            int off = db + ((db >> 10) << 4);
            v16bf kb0 = load_bf16x16(krow0 + off);
            v16bf kb1 = load_bf16x16(krow1 + off);
            #pragma unroll
            for (int f = 0; f < 8; ++f) {
                v16bf nb0 = kb0, nb1 = kb1;
                if (f < 7) {
                    db  = kdb + (f + 1) * 64;
                    off = db + ((db >> 10) << 4);        // +16B pad per 1024B
                    nb0 = load_bf16x16(krow0 + off);
                    nb1 = load_bf16x16(krow1 + off);
                }
                s0 = wmma_bf16(qa[f], kb0, s0);
                s1 = wmma_bf16(qa[f], kb1, s1);
                kb0 = nb0; kb1 = nb1;
            }
        }

        // ---- cross-wave (D-slice) reduction ----
        #pragma unroll
        for (int r = 0; r < 8; ++r) {
            sred[qs][0][ws][r + 8 * hi][col] = s0[r];
            sred[qs][1][ws][r + 8 * hi][col] = s1[r];
        }
        __syncthreads();
        float sf0[8], sf1[8];
        #pragma unroll
        for (int r = 0; r < 8; ++r) {
            const int row = r + 8 * hi;
            sf0[r] = sred[qs][0][0][row][col] + sred[qs][0][1][row][col] + sred[qs][0][2][row][col] + sred[qs][0][3][row][col];
            sf1[r] = sred[qs][1][0][row][col] + sred[qs][1][1][row][col] + sred[qs][1][2][row][col] + sred[qs][1][3][row][col];
        }
        __syncthreads();

        // ---- online masked softmax ----
        const float mk0 = (float)mk0i;
        const float mk1 = (float)mk1i;
        float alpha[8];
        #pragma unroll
        for (int r = 0; r < 8; ++r) {
            const float v0 = sf0[r] * mk0;
            const float v1 = sf1[r] * mk1;
            const float tmax = rmax16(fmaxf(v0, v1));
            const float mnew = fmaxf(m_run[r], tmax);
            alpha[r] = __expf(m_run[r] - mnew);
            const float p0 = mk0 * __expf(v0 - mnew);
            const float p1 = mk1 * __expf(v1 - mnew);
            l_run[r] = l_run[r] * alpha[r] + rsum16(p0 + p1);
            m_run[r] = mnew;
            pbuf[wave][r + 8 * hi][col]      = (__bf16)p0;
            pbuf[wave][r + 8 * hi][col + 16] = (__bf16)p1;
        }
        #pragma unroll
        for (int t = 0; t < 16; ++t)
            #pragma unroll
            for (int r = 0; r < 8; ++r) o_acc[t][r] *= alpha[r];

        // ---- P as A-fragment (wave-local LDS relayout) ----
        P2 pp;
        pp.a = *(const v8bf*)&pbuf[wave][col][hi * 16];
        pp.b = *(const v8bf*)&pbuf[wave][col][hi * 16 + 8];
        const v16bf pa = __builtin_bit_cast(v16bf, pp);

        // ---- O(slice) += P . V(slice) from LDS V tile (80B rows, pipelined frags) ----
        const char* vrow = smem + VOFF(cur) + (woff + col) * 80 + hi * 32;
        {
            v16bf vb = load_bf16x16(vrow);
            #pragma unroll
            for (int t = 0; t < 16; ++t) {
                v16bf nv = vb;
                if (t < 15) nv = load_bf16x16(vrow + (t + 1) * (16 * 80));
                o_acc[t] = wmma_bf16(pa, vb, o_acc[t]);
                vb = nv;
            }
        }
    }

    // ---- epilogue: normalize, apply query mask, store fp32 ----
    float scale[8];
    #pragma unroll
    for (int r = 0; r < 8; ++r) {
        const int row = q0 + r + 8 * hi;
        const float mq = (float)rep_mask[(size_t)b * L_ + row];
        const float l  = l_run[r];
        const float denom = l + ((l == 0.0f) ? 1.0f : 0.0f) + 1e-20f;
        scale[r] = mq / denom;
    }
    #pragma unroll
    for (int t = 0; t < 16; ++t) {
        #pragma unroll
        for (int r = 0; r < 8; ++r) {
            const size_t row = (size_t)(b * L_ + q0 + r + 8 * hi);
            out[row * D_ + woff + t * 16 + col] = o_acc[t][r] * scale[r];
        }
    }
}

extern "C" void kernel_launch(void* const* d_in, const int* in_sizes, int n_in,
                              void* d_out, int out_size, void* d_ws, size_t ws_size,
                              hipStream_t stream) {
    const float* q        = (const float*)d_in[0];
    const float* k        = (const float*)d_in[1];
    const float* v        = (const float*)d_in[2];
    const int*   rep_mask = (const int*)d_in[3];
    float*       out      = (float*)d_out;

    __bf16* kbf = (__bf16*)d_ws;                       // [B][L][D] bf16
    __bf16* vtr = kbf + (size_t)B_ * L_ * D_;          // [B][D][L] bf16

    const int n4 = B_ * L_ * D_ / 4;
    cvt_k_kernel<<<(n4 + 255) / 256, 256, 0, stream>>>(k, kbf, n4);
    transpose_v_kernel<<<dim3(L_ / 32, D_ / 32, B_), dim3(32, 8), 0, stream>>>(v, vtr);
    attn_kernel<<<dim3(L_ / 32, B_), 256, DYN_LDS, stream>>>(q, kbf, vtr, rep_mask, out);
}
